// RNN_33887291965934
// MI455X (gfx1250) — compile-verified, matrix operand order checked by reading, not run on hardware
//
#include <hip/hip_runtime.h>
#include <hip/hip_bf16.h>

// ---------------------------------------------------------------------------
// RNN:  xT = x @ xW + xB  (8192x2048 @ 2048x2048, bf16 WMMA, f32 accum)
//       h_t = tanh(xT_t + h_{t-1} @ hW + hB)   (sequential scan, 8192 steps)
//
// MI455X strategy:
//  * GEMM phase: LDS-tiled bf16 WMMA GEMM (128x64 block tile, BK=64 so each
//    LDS stage feeds 8 WMMAs/wave between barriers), packed bf16 converts.
//  * Scan phase: persistent kernel, hW pinned in LDS distributed across 64
//    workgroups (32 cols x 2048 K bf16 = 128 KB of the 320 KB WGP LDS),
//    hW B-fragments held in VGPRs for the whole scan.  Per step only the
//    8 KB h vector moves through L2.  d_out serves in place as both the
//    xT buffer and the h_t history.
// ---------------------------------------------------------------------------

typedef __attribute__((ext_vector_type(16))) __bf16 v16bf;
typedef __attribute__((ext_vector_type(8)))  __bf16 v8bf;
typedef __attribute__((ext_vector_type(8)))  float  v8f;

#define T_DIM 8192
#define D_DIM 2048
#define N_DIM 2048

// GEMM tiling
#define BM 128
#define BN 64
#define BK 64

// Scan config
#define SB        64     // persistent blocks
#define SCOLS     32     // output columns owned per block (64*32 = 2048)
#define STHREADS  512    // 16 waves: 2 column tiles x 8 K-ranges of 256

__device__ __forceinline__ unsigned short f2bf_bits(float f) {
  union { float f; unsigned u; } v; v.f = f;
  unsigned u = v.u;
  unsigned r = u + 0x7FFFu + ((u >> 16) & 1u);   // round-to-nearest-even
  return (unsigned short)(r >> 16);
}

// Packed f32x2 -> bf16x2 (single v_cvt_pk_bf16_f32 on gfx1250).
__device__ __forceinline__ unsigned pack_bf16x2(float a, float b) {
#if __has_builtin(__builtin_amdgcn_cvt_pk_bf16_f32)
  auto v = __builtin_amdgcn_cvt_pk_bf16_f32(a, b);
  return __builtin_bit_cast(unsigned, v);
#else
  return (unsigned)f2bf_bits(a) | ((unsigned)f2bf_bits(b) << 16);
#endif
}

// Build a 16x32 bf16 WMMA fragment from two contiguous 8-element (16B) LDS
// chunks.  Per the CDNA5 16-bit A/B layout, each lane's v16 holds
// K = 8*half + 0..7 in elements 0..7 and K = 16 + 8*half + 0..7 in 8..15.
__device__ __forceinline__ v16bf make_frag(const unsigned short* p0,
                                           const unsigned short* p1) {
  v8bf lo = *reinterpret_cast<const v8bf*>(p0);
  v8bf hi = *reinterpret_cast<const v8bf*>(p1);
  v16bf a;
#pragma unroll
  for (int i = 0; i < 8; ++i) { a[i] = lo[i]; a[i + 8] = hi[i]; }
  return a;
}

// ---------------------------------------------------------------------------
// Phase 1: d_out = x @ xW + xB   (f32 out, bf16 WMMA math)
// ---------------------------------------------------------------------------
__global__ __launch_bounds__(256) void gemm_xw_kernel(
    const float* __restrict__ x, const float* __restrict__ xW,
    const float* __restrict__ xB, float* __restrict__ out)
{
  __shared__ __align__(16) unsigned short As[BM][BK];  // A tile, [m][k]
  __shared__ __align__(16) unsigned short Bs[BN][BK];  // B tile transposed, [n][k]

  const int tid  = threadIdx.x;
  const int lane = tid & 31;
  const int wave = tid >> 5;
  const int half = lane >> 4;
  const int l16  = lane & 15;

  const int bx = blockIdx.x & 31;   // 32 N tiles
  const int by = blockIdx.x >> 5;   // 64 M tiles
  const int m0 = by * BM;
  const int n0 = bx * BN;

  const int wm = wave & 3;          // 4 slabs of 32 rows
  const int wn = wave >> 2;         // 2 slabs of 32 cols

  v8f acc[2][2];
#pragma unroll
  for (int i = 0; i < 2; ++i)
#pragma unroll
    for (int j = 0; j < 2; ++j)
#pragma unroll
      for (int e = 0; e < 8; ++e) acc[i][j][e] = 0.0f;

  for (int k0 = 0; k0 < D_DIM; k0 += BK) {
    // Stage A: 128x64 f32 -> bf16 row-major.  Each thread: 8 float4,
    // packed-converted to two b32 LDS stores each.
#pragma unroll
    for (int i = 0; i < 8; ++i) {
      int idx = tid + 256 * i;                 // 0..2047
      int row = idx >> 4;                      // 16 float4 per row
      int c4  = (idx & 15) << 2;
      const float4 v = *reinterpret_cast<const float4*>(
          &x[(size_t)(m0 + row) * D_DIM + k0 + c4]);
      *reinterpret_cast<unsigned*>(&As[row][c4 + 0]) = pack_bf16x2(v.x, v.y);
      *reinterpret_cast<unsigned*>(&As[row][c4 + 2]) = pack_bf16x2(v.z, v.w);
    }
    // Stage B: 64x64 f32, transposed into Bs[n][k].  Read k-row pairs so the
    // packed (k,k+1) bf16 pair is contiguous in the [n][k] layout.
#pragma unroll
    for (int i = 0; i < 2; ++i) {
      int idx = tid + 256 * i;                 // 0..511
      int k   = (idx >> 4) << 1;               // 0,2,..,62
      int c4  = (idx & 15) << 2;               // 0..60
      const float4 a = *reinterpret_cast<const float4*>(
          &xW[(size_t)(k0 + k + 0) * N_DIM + n0 + c4]);
      const float4 b = *reinterpret_cast<const float4*>(
          &xW[(size_t)(k0 + k + 1) * N_DIM + n0 + c4]);
      *reinterpret_cast<unsigned*>(&Bs[c4 + 0][k]) = pack_bf16x2(a.x, b.x);
      *reinterpret_cast<unsigned*>(&Bs[c4 + 1][k]) = pack_bf16x2(a.y, b.y);
      *reinterpret_cast<unsigned*>(&Bs[c4 + 2][k]) = pack_bf16x2(a.z, b.z);
      *reinterpret_cast<unsigned*>(&Bs[c4 + 3][k]) = pack_bf16x2(a.w, b.w);
    }
    // Prefetch next K tile into L2/WGP$ (no double buffering).
    if (k0 + BK < D_DIM) {
      __builtin_prefetch(
          &x[(size_t)(m0 + (tid >> 4)) * D_DIM + k0 + BK + ((tid & 15) << 2)], 0, 1);
      __builtin_prefetch(
          &xW[(size_t)(k0 + BK + ((tid >> 4) << 1)) * N_DIM + n0 + ((tid & 15) << 2)], 0, 1);
    }
    __syncthreads();

#pragma unroll
    for (int kc = 0; kc < BK; kc += 32) {
      v16bf afrag[2], bfrag[2];
#pragma unroll
      for (int mi = 0; mi < 2; ++mi) {
        int row = wm * 32 + mi * 16 + l16;
        afrag[mi] = make_frag(&As[row][kc + 8 * half], &As[row][kc + 16 + 8 * half]);
      }
#pragma unroll
      for (int ni = 0; ni < 2; ++ni) {
        int col = wn * 32 + ni * 16 + l16;
        bfrag[ni] = make_frag(&Bs[col][kc + 8 * half], &Bs[col][kc + 16 + 8 * half]);
      }
#pragma unroll
      for (int mi = 0; mi < 2; ++mi)
#pragma unroll
        for (int ni = 0; ni < 2; ++ni)
          acc[mi][ni] = __builtin_amdgcn_wmma_f32_16x16x32_bf16(
              false, afrag[mi], false, bfrag[ni], (short)0, acc[mi][ni],
              false, false);
    }
    __syncthreads();
  }

  // Epilogue: C layout = lane(col), VGPR r -> row r + 8*half.  Add bias.
#pragma unroll
  for (int ni = 0; ni < 2; ++ni) {
    int col = n0 + wn * 32 + ni * 16 + l16;
    float bias = xB[col];
#pragma unroll
    for (int mi = 0; mi < 2; ++mi) {
#pragma unroll
      for (int r = 0; r < 8; ++r) {
        int row = m0 + wm * 32 + mi * 16 + half * 8 + r;
        out[(size_t)row * N_DIM + col] = acc[mi][ni][r] + bias;
      }
    }
  }
}

// ---------------------------------------------------------------------------
// Counter reset (replay-deterministic grid barrier)
// ---------------------------------------------------------------------------
__global__ void zero_ctr_kernel(unsigned* ctr) { *ctr = 0u; }

// ---------------------------------------------------------------------------
// Phase 2: persistent scan.  Block b owns output columns [32b, 32b+32).
// hW slice lives in LDS as bf16 [c][k]; B fragments live in VGPRs.
// d_out row t holds xT_t before the step and h_t after it.
// ---------------------------------------------------------------------------
__global__ __launch_bounds__(STHREADS) void rnn_scan_kernel(
    const float* __restrict__ hW, const float* __restrict__ hB,
    float* __restrict__ out, unsigned* __restrict__ ctr)
{
  extern __shared__ __align__(16) unsigned char smem[];
  unsigned short* HWs = reinterpret_cast<unsigned short*>(smem); // [SCOLS][2048]
  unsigned short* hs  = HWs + SCOLS * 2048;                      // [2048] bf16 h
  float* partial      = reinterpret_cast<float*>(hs + 2048);     // [SCOLS]

  const int tid  = threadIdx.x;
  const int lane = tid & 31;
  const int wave = tid >> 5;
  const int half = lane >> 4;
  const int l16  = lane & 15;
  const int c0   = blockIdx.x * SCOLS;

  // Stage this block's hW slice once: HWs[c][k] = bf16(hW[k][c0+c]).
  // Read k-pairs so converts pack into single b32 LDS stores.
  {
    int c  = tid & (SCOLS - 1);
    int kb = (tid >> 5) * 128;         // 16 wave-groups x 128 k = 2048
    for (int kk = 0; kk < 128; kk += 2) {
      float v0 = hW[(size_t)(kb + kk + 0) * N_DIM + c0 + c];
      float v1 = hW[(size_t)(kb + kk + 1) * N_DIM + c0 + c];
      *reinterpret_cast<unsigned*>(&HWs[c * 2048 + kb + kk]) = pack_bf16x2(v0, v1);
    }
  }
  __syncthreads();

  // Wave roles: nt = 16-col tile (0..1), kr = K-range of 256 (0..7)
  const int nt = wave & 1;
  const int kr = wave >> 1;
  const int kw = kr * 256;
  const int bcol = nt * 16 + l16;

  // Preload 8 B fragments (hW) into VGPRs for the entire 8192-step scan.
  v16bf bfrag[8];
#pragma unroll
  for (int j = 0; j < 8; ++j) {
    int kb = kw + j * 32;
    bfrag[j] = make_frag(&HWs[bcol * 2048 + kb + 8 * half],
                         &HWs[bcol * 2048 + kb + 16 + 8 * half]);
  }

  float hb = 0.0f;
  if (tid < SCOLS) hb = hB[c0 + tid];

  const unsigned nblk = gridDim.x;

  for (int t = 0; t < T_DIM; ++t) {
    // h_{t-1} (zeros at t==0) -> LDS as bf16.  512 threads x 4 elems.
    {
      int i0 = tid * 4;
      uint2 p;
      if (t == 0) {
        p.x = 0u; p.y = 0u;
      } else {
        const float4 v = *reinterpret_cast<const float4*>(
            &out[(size_t)(t - 1) * N_DIM + i0]);
        p.x = pack_bf16x2(v.x, v.y);
        p.y = pack_bf16x2(v.z, v.w);
      }
      *reinterpret_cast<uint2*>(&hs[i0]) = p;
    }
    // partial = xT_t + hB  (read d_out row t before it is overwritten)
    if (tid < SCOLS) partial[tid] = out[(size_t)t * N_DIM + c0 + tid] + hb;
    if (t + 1 < T_DIM)  // warm L2/WGP$ for next step's xT row
      __builtin_prefetch(&out[(size_t)(t + 1) * N_DIM + c0], 0, 1);
    __syncthreads();

    // Each wave: 8 chained bf16 WMMAs over its 256-wide K range.
    // A fragment rows are all equal to h, so D row 0 = h @ hW slice.
    v8f acc;
#pragma unroll
    for (int e = 0; e < 8; ++e) acc[e] = 0.0f;
#pragma unroll
    for (int j = 0; j < 8; ++j) {
      int kb = kw + j * 32;
      v16bf a = make_frag(&hs[kb + 8 * half], &hs[kb + 16 + 8 * half]);
      acc = __builtin_amdgcn_wmma_f32_16x16x32_bf16(
          false, a, false, bfrag[j], (short)0, acc, false, false);
    }
    if (half == 0)                       // lanes 0..15: acc[0] = (M=0, N=l16)
      atomicAdd(&partial[nt * 16 + l16], acc[0]);   // ds_add_f32
    __syncthreads();

    // Activation + publish h_t (in place over xT_t)
    if (tid < SCOLS)
      out[(size_t)t * N_DIM + c0 + tid] = tanhf(partial[tid]);

    // Grid-wide barrier: monotonic counter, release store / acquire spin.
    __threadfence();
    if (tid == 0) {
      atomicAdd(ctr, 1u);
      const unsigned target = nblk * (unsigned)(t + 1);
      while (__hip_atomic_load(ctr, __ATOMIC_ACQUIRE,
                               __HIP_MEMORY_SCOPE_AGENT) < target) {
        __builtin_amdgcn_s_sleep(1);
      }
    }
    __syncthreads();
  }
}

// ---------------------------------------------------------------------------
extern "C" void kernel_launch(void* const* d_in, const int* in_sizes, int n_in,
                              void* d_out, int out_size, void* d_ws, size_t ws_size,
                              hipStream_t stream) {
  (void)in_sizes; (void)n_in; (void)out_size; (void)ws_size;
  const float* x  = (const float*)d_in[0];
  const float* xW = (const float*)d_in[1];
  const float* xB = (const float*)d_in[2];
  const float* hW = (const float*)d_in[3];
  const float* hB = (const float*)d_in[4];
  float* out    = (float*)d_out;
  unsigned* ctr = (unsigned*)d_ws;

  zero_ctr_kernel<<<1, 1, 0, stream>>>(ctr);

  // Phase 1: xT = x@xW + xB into d_out.  2048 blocks of 128x64 tiles.
  gemm_xw_kernel<<<dim3((T_DIM / BM) * (N_DIM / BN)), 256, 0, stream>>>(
      x, xW, xB, out);

  // Phase 2: persistent scan.  Dynamic LDS: hW slice + h + partials.
  size_t smem = (size_t)SCOLS * 2048 * sizeof(unsigned short)  // 128 KB hW
              + 2048 * sizeof(unsigned short)                  // 4 KB h
              + SCOLS * sizeof(float);                         // partials
  rnn_scan_kernel<<<dim3(SB), STHREADS, smem, stream>>>(hW, hB, out, ctr);
}
